// GCN_44495861187038
// MI455X (gfx1250) — matrix operand order repaired and checked by reading, not compiled, not run on hardware
//
#include <hip/hip_runtime.h>
#include <hip/hip_bf16.h>

// ---------------- problem constants ----------------
#define BDIM 32
#define CDIM 32
#define VDIM 2048
#define LDIM 12
#define SDIM 2
#define COUT 32
#define MROWS (BDIM * CDIM * LDIM) // 12288

typedef __bf16 bf16_t;
typedef __attribute__((ext_vector_type(16))) __bf16        v16bf;
typedef __attribute__((ext_vector_type(8)))  float         v8f;
typedef __attribute__((ext_vector_type(8)))  unsigned int  v8u;
typedef __attribute__((ext_vector_type(4)))  int           v4i;

// ---- CDNA5 async global->LDS path (guarded; fallback = VGPR staging) ----
#if defined(__has_builtin)
#  if __has_builtin(__builtin_amdgcn_global_load_async_to_lds_b128)
#    define HAS_ASYNC_LDS 1
#  endif
#endif
#ifndef HAS_ASYNC_LDS
#  define HAS_ASYNC_LDS 0
#endif

typedef __attribute__((address_space(1))) v4i* gas1_v4p; // "int4 __device__*"
typedef __attribute__((address_space(3))) v4i* las3_v4p; // "int4 __shared__*"

static __device__ __forceinline__ gas1_v4p as_global4(const void* p) {
  return (gas1_v4p)(unsigned long long)p;
}
static __device__ __forceinline__ las3_v4p as_lds4(void* p) {
  // flat LDS addresses: low 32 bits are the LDS offset (ISA 10.2 aperture rules)
  return (las3_v4p)(unsigned int)(unsigned long long)p;
}
static __device__ __forceinline__ void wait_async0() {
#if defined(__has_builtin) && __has_builtin(__builtin_amdgcn_s_wait_asynccnt)
  __builtin_amdgcn_s_wait_asynccnt(0);
#else
  asm volatile("s_wait_asynccnt 0x0" ::: "memory");
#endif
}

static __device__ __forceinline__ unsigned short f2bf(float f) {
  unsigned int u = __builtin_bit_cast(unsigned int, f);
  u += 0x7fffu + ((u >> 16) & 1u); // round-to-nearest-even
  return (unsigned short)(u >> 16);
}

static __device__ __forceinline__ v16bf frag_ld(const unsigned int* p) {
  return __builtin_bit_cast(v16bf, *(const v8u*)p);
}

// ---------------------------------------------------------------------------
// pack supports fp32 [S,V,V] -> bf16 in tiled WMMA B-fragment layout:
//   [s][tk = k/32][tn = n/16][256 dwords], dword (hn*8+j) = elements
//   (k = 32*tk + 8*(hn>>4) + 2*(j&3) + 16*(j>>2), n = 16*tn + (hn&15)) and (k+1, n).
// One 256-dword tile per block; done once, consumed 4x by the GEMMs with a
// pure identity async copy (adjacent tn tiles are contiguous -> 8KB/B-tile).
// ---------------------------------------------------------------------------
__global__ __launch_bounds__(256) void pack_support_kernel(
    const float* __restrict__ src, unsigned int* __restrict__ dst) {
  const int tid = threadIdx.x;
  int t = blockIdx.x;
  const int tn = t & (VDIM / 16 - 1); t >>= 7; // V/16 = 128
  const int tk = t & (VDIM / 32 - 1); t >>= 6; // V/32 = 64
  const int s  = t;
  const int hn   = tid >> 3;   // (half<<4) + n0
  const int j    = tid & 7;
  const int half = hn >> 4;
  const int n0   = hn & 15;
  const int k    = tk * 32 + half * 8 + 2 * (j & 3) + ((j >> 2) << 4);
  const int n    = tn * 16 + n0;
  const float* base = src + (size_t)s * VDIM * VDIM;
  unsigned int lo = f2bf(base[(size_t)k * VDIM + n]);
  unsigned int hi = f2bf(base[(size_t)(k + 1) * VDIM + n]);
  dst[(size_t)blockIdx.x * 256 + tid] = lo | (hi << 16);
}

// ---------------------------------------------------------------------------
// pack x [B,C,V,L] fp32 -> Xt [m=(b*C+c)*L+l, v] bf16 via LDS transpose.
// Bulk fill is contiguous both sides -> async b128 copies when available.
// ---------------------------------------------------------------------------
__global__ __launch_bounds__(256) void pack_x_kernel(
    const float* __restrict__ x, unsigned short* __restrict__ Xt) {
  __shared__ alignas(16) float sX[512 * LDIM]; // 24 KB
  const int tid = threadIdx.x;
  const int bc  = blockIdx.x >> 2;
  const int v0  = (blockIdx.x & 3) * 512;

  const size_t base = ((size_t)bc * VDIM + v0) * LDIM;
#if HAS_ASYNC_LDS
#pragma unroll
  for (int it = 0; it < 6; ++it) {
    int idx = (it << 10) + (tid << 2); // dword index, 16B-granular
    __builtin_amdgcn_global_load_async_to_lds_b128(
        as_global4(x + base + idx), as_lds4(&sX[idx]), 0, 0);
  }
  wait_async0();
#else
#pragma unroll
  for (int it = 0; it < 24; ++it) {
    int idx = tid + (it << 8);
    sX[idx] = x[base + idx];
  }
#endif
  __syncthreads();

  unsigned int* d32 = (unsigned int*)Xt;
#pragma unroll
  for (int it = 0; it < 12; ++it) {
    int q  = tid + (it << 8);      // 0..3071
    int l  = q >> 8;               // 0..11
    int vp = q & 255;              // v pair index
    int v  = vp << 1;
    unsigned int lo = f2bf(sX[v * LDIM + l]);
    unsigned int hi = f2bf(sX[(v + 1) * LDIM + l]);
    size_t row = (size_t)bc * LDIM + l;
    d32[row * (VDIM / 2) + ((v0 + v) >> 1)] = lo | (hi << 16);
  }
}

// ---------------------------------------------------------------------------
// bf16 GEMM: D[M,N] = A[M,K] * Bsw[K,N], f32 accumulate via
// v_wmma_f32_16x16x32_bf16. A is row-major; Bsw is pre-swizzled tiled
// B-fragment layout (see pack_support_kernel). Tiles 128x128x32,
// 256 threads = 8 waves (4x2), double-buffered LDS. Both operand tiles are
// staged by GLOBAL_LOAD_ASYNC_TO_LDS_B128 (ASYNCcnt), zero repack VALU.
// ---------------------------------------------------------------------------
__global__ __launch_bounds__(256) void gemm_bf16_kernel(
    const unsigned short* __restrict__ A,   // M x K bf16 row-major
    const unsigned int*   __restrict__ Bsw, // [K/32][N/16][256] dwords
    unsigned short* __restrict__ D,         // M x N bf16 row-major
    int Mdim, int Ndim, int Kdim) {
  __shared__ alignas(32) unsigned int sA[2][2048]; // 128x32 bf16, swizzled
  __shared__ alignas(32) unsigned int sB[2][2048]; // 32x128 bf16, swizzled
  (void)Mdim;

  const int tid  = threadIdx.x;
  const int lane = tid & 31;
  const int wid  = tid >> 5;
  const int wm   = wid & 3;  // wave row (32 rows each)
  const int wn   = wid >> 2; // wave col (64 cols each)
  const int bn   = blockIdx.x * 128;
  const int bm   = blockIdx.y * 128;

  // ---- per-thread A staging geometry: (row m, half h) -> two 16B chunks ----
  const int m  = tid >> 1;
  const int mm = m & 15;
  const int h  = tid & 1;
  const unsigned short* arow = A + (size_t)(bm + m) * Kdim; // + k0 later
  const int aidx1 = ((m >> 4) << 8) + (mm << 3) + (h << 2);        // kp 8h..8h+3
  const int aidx2 = ((m >> 4) << 8) + ((16 + mm) << 3) + (h << 2); // kp 8h+4..8h+7

  // ---- B staging: identity copy, 8 contiguous dwords per thread ----
  const unsigned int* bbase = Bsw + (size_t)bn * 16 + (tid << 3); // + kt stride
  const size_t bstride = (size_t)Ndim << 4; // dwords per K-tile row of tiles

  v8f acc[2][4] = {};
  const int KT = Kdim >> 5;

  auto stage = [&](int kt, int buf) {
#if HAS_ASYNC_LDS
    const char* ga = (const char*)(arow + (kt << 5)) + (h << 5);
    __builtin_amdgcn_global_load_async_to_lds_b128(
        as_global4(ga), as_lds4(&sA[buf][aidx1]), 0, 0);
    __builtin_amdgcn_global_load_async_to_lds_b128(
        as_global4(ga + 16), as_lds4(&sA[buf][aidx2]), 0, 0);
    const unsigned int* gb = bbase + (size_t)kt * bstride;
    __builtin_amdgcn_global_load_async_to_lds_b128(
        as_global4(gb), as_lds4(&sB[buf][tid << 3]), 0, 0);
    __builtin_amdgcn_global_load_async_to_lds_b128(
        as_global4(gb + 4), as_lds4(&sB[buf][(tid << 3) + 4]), 0, 0);
#else
    {
      const unsigned int* gsrc = (const unsigned int*)(arow + (kt << 5));
      int kpb = h << 3;
#pragma unroll
      for (int q = 0; q < 8; ++q) {
        int kp = kpb + q;
        int hf = (kp >> 2) & 1;
        int j  = (kp & 3) + ((kp >> 3) << 2);
        sA[buf][((m >> 4) << 8) + (((hf << 4) + mm) << 3) + j] = gsrc[kp];
      }
      const unsigned int* gb = bbase + (size_t)kt * bstride;
#pragma unroll
      for (int q = 0; q < 8; ++q) sB[buf][(tid << 3) + q] = gb[q];
    }
#endif
  };

  stage(0, 0); // prologue
#if HAS_ASYNC_LDS
  wait_async0();
#endif
  __syncthreads();

  for (int kt = 0; kt < KT; ++kt) {
    const int cur = kt & 1;
    if (kt + 1 < KT) stage(kt + 1, cur ^ 1);

    const unsigned int* pa = &sA[cur][((wm * 2) << 8) + lane * 8];
    v16bf a0 = frag_ld(pa);
    v16bf a1 = frag_ld(pa + 256);
#pragma unroll
    for (int jn = 0; jn < 4; ++jn) {
      v16bf bf = frag_ld(&sB[cur][((wn * 4 + jn) << 8) + lane * 8]);
      acc[0][jn] = __builtin_amdgcn_wmma_f32_16x16x32_bf16(
          false, a0, false, bf, (short)0, acc[0][jn], false, false);
      acc[1][jn] = __builtin_amdgcn_wmma_f32_16x16x32_bf16(
          false, a1, false, bf, (short)0, acc[1][jn], false, false);
    }
#if HAS_ASYNC_LDS
    wait_async0(); // next tile's async DMA landed
#endif
    __syncthreads(); // everyone done reading `cur` before it is overwritten
  }

  // ---- epilogue: f32 acc -> bf16 stores (16-lane contiguous 32B runs) ----
  const int rsel = (lane >> 4) << 3; // +8 rows for upper half-wave
  const int ncol = lane & 15;
#pragma unroll
  for (int i = 0; i < 2; ++i) {
    int rowb = bm + (wm * 2 + i) * 16 + rsel;
#pragma unroll
    for (int jn = 0; jn < 4; ++jn) {
      int col = bn + (wn * 4 + jn) * 16 + ncol;
#pragma unroll
      for (int r = 0; r < 8; ++r) {
        D[(size_t)(rowb + r) * Ndim + col] = f2bf(acc[i][jn][r]);
      }
    }
  }
}

// ---------------------------------------------------------------------------
// Channel mix: per (b,l,v-tile of 128): Y'[o,v] = sum_k Wk[32x32]*Hk[32x128]+bias
// K = 5 blocks of 32 (x, A0x, A0^2x, A1x, A1^2x). Output layout [b,o,l,v]
// (v contiguous -> coalesced f32 stores). 128 threads = 4 waves.
// ---------------------------------------------------------------------------
__global__ __launch_bounds__(128) void mix_kernel(
    const unsigned short* __restrict__ Xt,  // [MROWS, V] bf16
    const unsigned short* __restrict__ Hb,  // 4 x [MROWS, V] bf16
    const float* __restrict__ W,            // [32, 160]
    const float* __restrict__ bias,         // [32]
    float* __restrict__ Yp) {               // [B, O, L, V] f32
  __shared__ alignas(32) unsigned int sW[5 * 512];  // 5 A-tiles 32x32
  __shared__ alignas(32) unsigned int sH[5 * 2048]; // 5 B-tiles 32x128

  const int tid  = threadIdx.x;
  const int lane = tid & 31;
  const int wid  = tid >> 5;
  const int vt   = blockIdx.x & 15;
  const int l    = (blockIdx.x >> 4) % LDIM;
  const int b    = blockIdx.x / (16 * LDIM);
  const int v0   = vt * 128;

  // stage W (fp32 -> bf16, swizzled A-fragment layout), 2560 dwords
  for (int it = 0; it < 20; ++it) {
    int item = tid + (it << 7);
    int k  = item >> 9;        // source block 0..4
    int rm = item & 511;
    int o  = rm >> 4;          // 0..31
    int cp = rm & 15;          // c pair 0..15
    unsigned int lo = f2bf(W[o * 160 + k * 32 + 2 * cp]);
    unsigned int hi = f2bf(W[o * 160 + k * 32 + 2 * cp + 1]);
    int half = (cp >> 2) & 1;
    int j    = (cp & 3) + ((cp >> 3) << 2);
    sW[k * 512 + ((o >> 4) << 8) + (((half << 4) + (o & 15)) << 3) + j] = lo | (hi << 16);
  }
  // stage H/X tiles (bf16, repack to B-fragment layout), 5120 items
  for (int it = 0; it < 40; ++it) {
    int item = tid + (it << 7);
    int k    = item >> 10;     // source block 0..4
    int rm   = item & 1023;
    int kp2  = rm >> 6;        // c pair 0..15
    int np   = rm & 63;        // n pair 0..63
    int c    = kp2 << 1;
    int n    = np << 1;
    const unsigned short* src =
        (k == 0) ? Xt : (Hb + (size_t)(k - 1) * MROWS * VDIM);
    size_t row0 = ((size_t)(b * CDIM + c) * LDIM + l) * VDIM + v0 + n;
    unsigned int a = *(const unsigned int*)(src + row0);
    unsigned int d = *(const unsigned int*)(src + row0 + (size_t)LDIM * VDIM);
    unsigned int d0 = (a & 0xffffu) | (d << 16);
    unsigned int d1 = (a >> 16) | (d & 0xffff0000u);
    int half = (kp2 >> 2) & 1;
    int j    = (kp2 & 3) + ((kp2 >> 3) << 2);
    int addr = k * 2048 + ((n >> 4) << 8) + (((half << 4) + (n & 15)) << 3) + j;
    sH[addr]     = d0;
    sH[addr + 8] = d1;
  }
  __syncthreads();

  v8f acc[2][2] = {};
#pragma unroll
  for (int k = 0; k < 5; ++k) {
    const unsigned int* pa = sW + k * 512 + lane * 8;
    v16bf a0 = frag_ld(pa);
    v16bf a1 = frag_ld(pa + 256);
#pragma unroll
    for (int jn = 0; jn < 2; ++jn) {
      v16bf bf = frag_ld(sH + k * 2048 + ((wid * 2 + jn) << 8) + lane * 8);
      acc[0][jn] = __builtin_amdgcn_wmma_f32_16x16x32_bf16(
          false, a0, false, bf, (short)0, acc[0][jn], false, false);
      acc[1][jn] = __builtin_amdgcn_wmma_f32_16x16x32_bf16(
          false, a1, false, bf, (short)0, acc[1][jn], false, false);
    }
  }

  const int rsel = (lane >> 4) << 3;
  const int ncol = lane & 15;
#pragma unroll
  for (int i = 0; i < 2; ++i) {
#pragma unroll
    for (int jn = 0; jn < 2; ++jn) {
#pragma unroll
      for (int r = 0; r < 8; ++r) {
        int o = i * 16 + rsel + r;
        int v = v0 + (wid * 2 + jn) * 16 + ncol;
        float val = acc[i][jn][r] + bias[o];
        Yp[((size_t)(b * COUT + o) * LDIM + l) * VDIM + v] = val;
      }
    }
  }
}

// ---------------------------------------------------------------------------
// Y'[b,o,l,v] -> y[b,o,v,l] via LDS tile; both sides fully coalesced
// (writes emit contiguous 12-float runs over consecutive v).
// ---------------------------------------------------------------------------
__global__ __launch_bounds__(256) void untranspose_kernel(
    const float* __restrict__ Yp, float* __restrict__ y) {
  __shared__ float sT[LDIM * 257];
  const int tid = threadIdx.x;
  const int bo  = blockIdx.x >> 3;         // (b*32+o)
  const int v0  = (blockIdx.x & 7) * 256;

#pragma unroll
  for (int it = 0; it < LDIM; ++it) {
    sT[it * 257 + tid] = Yp[((size_t)bo * LDIM + it) * VDIM + v0 + tid];
  }
  __syncthreads();

  const size_t base = ((size_t)bo * VDIM + v0) * LDIM;
#pragma unroll
  for (int it = 0; it < LDIM; ++it) {
    int q = tid + (it << 8);                // 0..3071 = v*12 + l
    int v = q / LDIM;
    int l = q % LDIM;
    y[base + q] = sT[l * 257 + v];
  }
}

// ---------------------------------------------------------------------------
extern "C" void kernel_launch(void* const* d_in, const int* in_sizes, int n_in,
                              void* d_out, int out_size, void* d_ws, size_t ws_size,
                              hipStream_t stream) {
  (void)in_sizes; (void)n_in; (void)out_size; (void)ws_size;
  const float* x        = (const float*)d_in[0];
  const float* supports = (const float*)d_in[1];
  const float* W        = (const float*)d_in[2];
  const float* bias     = (const float*)d_in[3];
  float* y = (float*)d_out;

  // workspace layout (bytes): A_swizzled_bf16 | Xt_bf16 | H[4]_bf16 | Y'(f32)
  char* ws = (char*)d_ws;
  unsigned int* Asw = (unsigned int*)ws;
  size_t off = (size_t)SDIM * VDIM * VDIM * 2;           // 16.8 MB
  unsigned short* Xt = (unsigned short*)(ws + off);
  off += (size_t)MROWS * VDIM * 2;                       // +50.3 MB
  unsigned short* Hb = (unsigned short*)(ws + off);
  off += (size_t)4 * MROWS * VDIM * 2;                   // +201 MB
  float* Yp = (float*)(ws + off);                        // +100.7 MB

  // 1) precision/layout packing (supports go straight to WMMA B-tile layout)
  pack_support_kernel<<<SDIM * (VDIM / 32) * (VDIM / 16), 256, 0, stream>>>(
      supports, Asw);
  pack_x_kernel<<<BDIM * CDIM * (VDIM / 512), 256, 0, stream>>>(x, Xt);

  // 2) graph propagation: four [12288x2048]x[2048x2048] bf16 GEMMs
  unsigned short* H0 = Hb;
  unsigned short* H1 = H0 + (size_t)MROWS * VDIM;
  unsigned short* H2 = H1 + (size_t)MROWS * VDIM;
  unsigned short* H3 = H2 + (size_t)MROWS * VDIM;
  const unsigned int* A0 = Asw;
  const unsigned int* A1 = Asw + (size_t)VDIM * VDIM / 2; // dwords per support
  dim3 ggrid(VDIM / 128, MROWS / 128);
  gemm_bf16_kernel<<<ggrid, 256, 0, stream>>>(Xt, A0, H0, MROWS, VDIM, VDIM);
  gemm_bf16_kernel<<<ggrid, 256, 0, stream>>>(H0, A0, H1, MROWS, VDIM, VDIM);
  gemm_bf16_kernel<<<ggrid, 256, 0, stream>>>(Xt, A1, H2, MROWS, VDIM, VDIM);
  gemm_bf16_kernel<<<ggrid, 256, 0, stream>>>(H2, A1, H3, MROWS, VDIM, VDIM);

  // 3) 1x1 conv channel mix (K=160 WMMA GEMM) -> [b,o,l,v]
  mix_kernel<<<BDIM * LDIM * (VDIM / 128), 128, 0, stream>>>(Xt, Hb, W, bias, Yp);

  // 4) layout fix -> [b,o,v,l]
  untranspose_kernel<<<BDIM * COUT * (VDIM / 256), 256, 0, stream>>>(Yp, y);
}
